// MultiTaskLSTM_1666447311225
// MI455X (gfx1250) — matrix operand order repaired and checked by reading, not compiled, not run on hardware
//
#include <hip/hip_runtime.h>
#include <hip/hip_bf16.h>
#include <math.h>

#define T_STEPS 1024
#define I_DIM   64
#define H_DIM   256
#define KTOT    320    // I + H, uniform K loop of 10 x 32
#define BT      32     // batch rows per workgroup (2 M-tiles)
#define NTHR    512    // 16 waves
#define NG      1024   // 4H gate columns

typedef __attribute__((ext_vector_type(16))) __bf16 v16bf;
typedef __attribute__((ext_vector_type(8)))  float  v8f;

union ABFrag {
  uint4 q[2];   // 32 bytes: 16 bf16 per lane
  v16bf v;
};

__device__ __forceinline__ unsigned short f2bf(float f) {
  unsigned u = __float_as_uint(f);
  u += 0x7FFFu + ((u >> 16) & 1u);     // round-to-nearest-even
  return (unsigned short)(u >> 16);
}

__device__ __forceinline__ float fast_tanh(float x) {
#if __has_builtin(__builtin_amdgcn_tanhf)
  return __builtin_amdgcn_tanhf(x);
#else
  return tanhf(x);
#endif
}
__device__ __forceinline__ float sigm(float x) {
  return 0.5f * fast_tanh(0.5f * x) + 0.5f;   // one trans op per sigmoid
}

// ---------------------------------------------------------------------------
// Prep: build W_cat[N=1024][K=320] bf16 (row n = [W_ih[n,:64] | W_hh[n,:256]])
// and combined bias b_ih + b_hh (f32[1024]).
// ---------------------------------------------------------------------------
__global__ __launch_bounds__(256) void lstm_prep_kernel(
    const float* __restrict__ Wih, const float* __restrict__ Whh,
    const float* __restrict__ bih, const float* __restrict__ bhh,
    unsigned short* __restrict__ Wcat, float* __restrict__ bias)
{
  int idx = blockIdx.x * 256 + threadIdx.x;
  if (idx < NG * KTOT) {
    int n = idx / KTOT, k = idx - n * KTOT;
    float v = (k < I_DIM) ? Wih[n * I_DIM + k] : Whh[n * H_DIM + (k - I_DIM)];
    Wcat[idx] = f2bf(v);
  }
  if (idx < NG) bias[idx] = bih[idx] + bhh[idx];
}

// ---------------------------------------------------------------------------
// Persistent LSTM: 16 WGs x 512 threads (16 waves), 32 batch rows per WG.
// Wave w owns hidden columns j = w*16 + 0..15: its 4 N-tiles are the i/f/g/o
// gate columns for exactly those hidden units, so the nonlinearity + cell
// update stay entirely in-wave. Each B fragment feeds 2 M-tiles (rows 0-15
// and 16-31) -> 1 WMMA per global_load_b128, half the L2 traffic of BT=16.
// ---------------------------------------------------------------------------
__global__ __launch_bounds__(NTHR) void lstm_main_kernel(
    const float* __restrict__ x,
    const unsigned short* __restrict__ Wcat,
    const float* __restrict__ bias,
    const float* __restrict__ Wdir, const float* __restrict__ bdir,
    const float* __restrict__ Wq,   const float* __restrict__ bq,
    const float* __restrict__ Wrr,  const float* __restrict__ brr,
    const float* __restrict__ Wsl,  const float* __restrict__ bsl,
    float* __restrict__ out)
{
  __shared__ unsigned short xh[BT * KTOT];   // [x_t | h] bf16, 20 KB
  __shared__ float hbuf[BT * H_DIM];         // final h f32 for heads, 32 KB

  const int tid  = threadIdx.x;
  const int wave = tid >> 5;      // 0..15
  const int lane = tid & 31;
  const int hi   = lane >> 4;     // lane half
  const int m    = lane & 15;     // A row / B-N / C col within tile
  const int b0   = blockIdx.x * BT;

  // zero h region + stage x_0
  for (int i = tid; i < BT * KTOT; i += NTHR) xh[i] = 0;
  __syncthreads();
  {
    int r = tid >> 4, i0 = (tid & 15) * 4;   // 32 rows x 16 quads
    float4 xv = *(const float4*)(x + ((size_t)(b0 + r) * T_STEPS + 0) * I_DIM + i0);
    uint2 p;
    p.x = (unsigned)f2bf(xv.x) | ((unsigned)f2bf(xv.y) << 16);
    p.y = (unsigned)f2bf(xv.z) | ((unsigned)f2bf(xv.w) << 16);
    *(uint2*)&xh[r * KTOT + i0] = p;
  }
  __syncthreads();

  // Loop-invariant per-lane constants: B row offsets and bias values per gate.
  int   nOff[4];
  float bv[4];
  #pragma unroll
  for (int g = 0; g < 4; ++g) {
    int n = g * 256 + wave * 16 + m;         // gate column for this lane
    nOff[g] = n * KTOT;
    bv[g]   = bias[n];
  }

  float cst0[8], cst1[8];                    // cell state (2 M-tiles), in regs
  #pragma unroll
  for (int v = 0; v < 8; ++v) { cst0[v] = 0.f; cst1[v] = 0.f; }

  const int kbA = hi * 8;                    // A frag K sub-offset per lane half
  const int kbB = hi * 8;                    // B frag K sub-offset per lane half

  #pragma unroll 1
  for (int t = 0; t < T_STEPS; ++t) {
    // --- gates = [x|h] @ Wcat^T + bias : k-outer, gate-inner, 2 M-tiles ---
    v8f acc0[4], acc1[4];
    #pragma unroll
    for (int g = 0; g < 4; ++g) {
      float b = bv[g];
      acc0[g] = (v8f){b, b, b, b, b, b, b, b};
      acc1[g] = acc0[g];
    }
    #pragma unroll 1
    for (int k = 0; k < 10; ++k) {
      ABFrag af0, af1;
      const unsigned short* ap0 = &xh[m * KTOT + k * 32 + kbA];
      const unsigned short* ap1 = ap0 + 16 * KTOT;     // rows 16..31
      af0.q[0] = *(const uint4*)(ap0);
      af0.q[1] = *(const uint4*)(ap0 + 16);
      af1.q[0] = *(const uint4*)(ap1);
      af1.q[1] = *(const uint4*)(ap1 + 16);
      #pragma unroll
      for (int g = 0; g < 4; ++g) {
        ABFrag bf;
        const unsigned short* wp = Wcat + nOff[g] + k * 32 + kbB;
        bf.q[0] = *(const uint4*)(wp);
        bf.q[1] = *(const uint4*)(wp + 16);
        acc0[g] = __builtin_amdgcn_wmma_f32_16x16x32_bf16(
            false, af0.v, false, bf.v, (short)0, acc0[g], false, false);
        acc1[g] = __builtin_amdgcn_wmma_f32_16x16x32_bf16(
            false, af1.v, false, bf.v, (short)0, acc1[g], false, false);
      }
    }
    __syncthreads();                          // all LDS reads of xh done

    // --- stage x_{t+1} (write phase) ---
    if (t + 1 < T_STEPS) {
      int r = tid >> 4, i0 = (tid & 15) * 4;
      float4 xv = *(const float4*)(x + ((size_t)(b0 + r) * T_STEPS + (t + 1)) * I_DIM + i0);
      uint2 p;
      p.x = (unsigned)f2bf(xv.x) | ((unsigned)f2bf(xv.y) << 16);
      p.y = (unsigned)f2bf(xv.z) | ((unsigned)f2bf(xv.w) << 16);
      *(uint2*)&xh[r * KTOT + i0] = p;
    }

    // --- gate nonlinearity + cell update, fully in-wave ---
    const int j = wave * 16 + m;             // hidden unit column
    #pragma unroll
    for (int v = 0; v < 8; ++v) {
      // M-tile 0: rows 0..15
      {
        float iv = sigm(acc0[0][v]);
        float fv = sigm(acc0[1][v]);
        float gv = fast_tanh(acc0[2][v]);
        float ov = sigm(acc0[3][v]);
        float c  = fv * cst0[v] + iv * gv;
        cst0[v] = c;
        float h  = ov * fast_tanh(c);
        int r = v + hi * 8;
        xh[r * KTOT + I_DIM + j] = f2bf(h);
        if (t == T_STEPS - 1) hbuf[r * H_DIM + j] = h;
      }
      // M-tile 1: rows 16..31
      {
        float iv = sigm(acc1[0][v]);
        float fv = sigm(acc1[1][v]);
        float gv = fast_tanh(acc1[2][v]);
        float ov = sigm(acc1[3][v]);
        float c  = fv * cst1[v] + iv * gv;
        cst1[v] = c;
        float h  = ov * fast_tanh(c);
        int r = 16 + v + hi * 8;
        xh[r * KTOT + I_DIM + j] = f2bf(h);
        if (t == T_STEPS - 1) hbuf[r * H_DIM + j] = h;
      }
    }
    __syncthreads();                          // h(t)/x(t+1) visible for next GEMM
  }

  // --- fused heads: 8 outputs per batch row (32 rows -> 256 threads) ---
  if (tid < 8 * BT) {
    int r = tid >> 3, k = tid & 7;
    const float* wrow; float bh; int mode;    // 0 tanh, 1 identity, 2 sigmoid
    if (k == 0)      { wrow = Wdir;                 bh = bdir[0];  mode = 0; }
    else if (k <= 5) { wrow = Wq + (k - 1) * H_DIM; bh = bq[k - 1]; mode = 1; }
    else if (k == 6) { wrow = Wrr;                  bh = brr[0];   mode = 1; }
    else             { wrow = Wsl;                  bh = bsl[0];   mode = 2; }
    float sacc = bh;
    for (int jj = 0; jj < H_DIM; ++jj) sacc += hbuf[r * H_DIM + jj] * wrow[jj];
    if (mode == 0)      sacc = fast_tanh(sacc);
    else if (mode == 2) sacc = sigm(sacc);
    out[(size_t)(b0 + r) * 8 + k] = sacc;
  }
}

// ---------------------------------------------------------------------------
extern "C" void kernel_launch(void* const* d_in, const int* in_sizes, int n_in,
                              void* d_out, int out_size, void* d_ws, size_t ws_size,
                              hipStream_t stream) {
  const float* x    = (const float*)d_in[0];
  const float* Wih  = (const float*)d_in[1];
  const float* Whh  = (const float*)d_in[2];
  const float* bih  = (const float*)d_in[3];
  const float* bhh  = (const float*)d_in[4];
  const float* Wdir = (const float*)d_in[5];
  const float* bdir = (const float*)d_in[6];
  const float* Wq   = (const float*)d_in[7];
  const float* bq   = (const float*)d_in[8];
  const float* Wrr  = (const float*)d_in[9];
  const float* brr  = (const float*)d_in[10];
  const float* Wsl  = (const float*)d_in[11];
  const float* bsl  = (const float*)d_in[12];
  float* out = (float*)d_out;

  unsigned short* Wcat = (unsigned short*)d_ws;                       // 640 KB
  float* bias = (float*)((char*)d_ws + (size_t)NG * KTOT * 2);        // 4 KB

  lstm_prep_kernel<<<(NG * KTOT + 255) / 256, 256, 0, stream>>>(
      Wih, Whh, bih, bhh, Wcat, bias);

  lstm_main_kernel<<<512 / BT, NTHR, 0, stream>>>(
      x, Wcat, bias, Wdir, bdir, Wq, bq, Wrr, brr, Wsl, bsl, out);
}
